// TopKSAE_3307124818299
// MI455X (gfx1250) — matrix operand order repaired and checked by reading, not compiled, not run on hardware
//
#include <hip/hip_runtime.h>

#define D_DIM 1024
#define N_DIM 16384
#define B_DIM 8192
#define K_TOP 64

typedef __attribute__((ext_vector_type(16))) _Float16 v16h;
typedef __attribute__((ext_vector_type(8)))  _Float16 v8h;
typedef __attribute__((ext_vector_type(8)))  float    v8f;

// ---------------------------------------------------------------------------
// Encoder: z = relu((x - b) @ W_enc^T)  via v_wmma_f32_16x16x32_f16
// Block tile 128x128, K-step 32, 256 threads = 8 wave32, wave owns 16x128.
// ---------------------------------------------------------------------------
#define BM 128
#define BN 128
#define BK 32
#define LDA (BK + 8)   // 40 halves = 80B row stride (16B aligned, conflict pad)

__launch_bounds__(256, 2)
__global__ void sae_encoder(const float* __restrict__ x,     // [B, D]
                            const float* __restrict__ W,     // [N, D] (W_enc)
                            const float* __restrict__ bias,  // [D]
                            float* __restrict__ z)           // [B, N]
{
  __shared__ _Float16 As[BM][LDA];
  __shared__ _Float16 Bs[BN][LDA];

  const int tid  = threadIdx.x;
  const int wave = tid >> 5;
  const int lane = tid & 31;
  const int brow = blockIdx.y * BM;   // batch rows
  const int bcol = blockIdx.x * BN;   // latent cols

  v8f zero = {0.f, 0.f, 0.f, 0.f, 0.f, 0.f, 0.f, 0.f};
  v8f acc[8];
#pragma unroll
  for (int i = 0; i < 8; ++i) acc[i] = zero;

  // A fragment addressing per ISA 7.12.2 (16-bit A 16x32):
  // lanes 0-15: M=lane,    K = {0..7, 16..23}
  // lanes 16-31: M=lane-16, K = {8..15, 24..31}
  const int arow = wave * 16 + (lane & 15);
  const int akb  = (lane >> 4) * 8;
  // B fragment (16-bit B 32x16): lanes 0-15: N=lane, K = {0..7, 16..23}; lanes 16-31: K = {8..15, 24..31}
  const int bkb  = (lane >> 4) * 8;

  for (int kk = 0; kk < D_DIM; kk += BK) {
    // cooperative load: 128x32 floats each for A and B, 16 floats/thread each
#pragma unroll
    for (int t = 0; t < 4; ++t) {
      int seg = tid + t * 256;          // 0..1023 float4 segments
      int r   = seg >> 3;               // 0..127
      int c   = (seg & 7) * 4;          // 0..28
      float4 bb = *(const float4*)(bias + kk + c);
      float4 xa = *(const float4*)(x + (size_t)(brow + r) * D_DIM + kk + c);
      As[r][c + 0] = (_Float16)(xa.x - bb.x);
      As[r][c + 1] = (_Float16)(xa.y - bb.y);
      As[r][c + 2] = (_Float16)(xa.z - bb.z);
      As[r][c + 3] = (_Float16)(xa.w - bb.w);
      float4 wa = *(const float4*)(W + (size_t)(bcol + r) * D_DIM + kk + c);
      Bs[r][c + 0] = (_Float16)wa.x;
      Bs[r][c + 1] = (_Float16)wa.y;
      Bs[r][c + 2] = (_Float16)wa.z;
      Bs[r][c + 3] = (_Float16)wa.w;
    }
    __syncthreads();

    v8h alo = *(const v8h*)&As[arow][akb];
    v8h ahi = *(const v8h*)&As[arow][akb + 16];
    v16h afrag = __builtin_shufflevector(alo, ahi,
        0, 1, 2, 3, 4, 5, 6, 7, 8, 9, 10, 11, 12, 13, 14, 15);

#pragma unroll
    for (int nt = 0; nt < 8; ++nt) {
      int bc = nt * 16 + (lane & 15);
      v8h blo = *(const v8h*)&Bs[bc][bkb];
      v8h bhi = *(const v8h*)&Bs[bc][bkb + 16];
      v16h bfrag = __builtin_shufflevector(blo, bhi,
          0, 1, 2, 3, 4, 5, 6, 7, 8, 9, 10, 11, 12, 13, 14, 15);
      acc[nt] = __builtin_amdgcn_wmma_f32_16x16x32_f16(
          false, afrag, false, bfrag, (short)0, acc[nt], false, false);
    }
    __syncthreads();
  }

  // C/D layout: VGPR v -> lanes 0-15: M=v, N=lane; lanes 16-31: M=v+8, N=lane-16
  const int mbase = (lane >> 4) * 8;
  const int ncol  = lane & 15;
#pragma unroll
  for (int nt = 0; nt < 8; ++nt) {
#pragma unroll
    for (int v = 0; v < 8; ++v) {
      float val = acc[nt][v];
      val = val > 0.0f ? val : 0.0f;   // fused ReLU
      size_t gr = (size_t)(brow + wave * 16 + mbase + v);
      size_t gc = (size_t)(bcol + nt * 16 + ncol);
      z[gr * N_DIM + gc] = val;
    }
  }
}

// ---------------------------------------------------------------------------
// Top-K: one WG per row; whole 64KB row cached in LDS (320KB/WGP).
// Exact 4-level radix select on float bits (all values >= 0 after ReLU,
// so uint order == float order). Masks row in place, emits compact pairs.
// ---------------------------------------------------------------------------
__launch_bounds__(256)
__global__ void sae_topk(float* __restrict__ z,       // [B, N] in/out
                         int* __restrict__ cidx,      // [B, K] or null
                         float* __restrict__ cval)    // [B, K] or null
{
  extern __shared__ unsigned int sm[];
  unsigned int* v    = sm;            // N_DIM row cache
  unsigned int* hist = sm + N_DIM;    // 256 bins
  unsigned int* suf  = hist + 256;    // 256 suffix sums
  unsigned int* misc = suf + 256;     // [0]=prefix [1]=need [2]=eqctr [3]=slot

  const int tid = threadIdx.x;
  const int row = blockIdx.x;
  const size_t rowbase = (size_t)row * N_DIM;
  const unsigned int* zu = (const unsigned int*)(z + rowbase);

  for (int i = tid; i < N_DIM; i += 256) v[i] = zu[i];
  if (tid == 0) { misc[0] = 0u; misc[1] = K_TOP; misc[2] = 0u; misc[3] = 0u; }
  __syncthreads();

  for (int level = 0; level < 4; ++level) {
    const int shift = 24 - 8 * level;
    const unsigned int pmask  = (level == 0) ? 0u : (0xFFFFFFFFu << (shift + 8));
    const unsigned int prefix = misc[0];
    const unsigned int need   = misc[1];
    hist[tid] = 0u;
    __syncthreads();
    for (int i = tid; i < N_DIM; i += 256) {
      unsigned int bv = v[i];
      if ((bv & pmask) == prefix)
        atomicAdd(&hist[(bv >> shift) & 255u], 1u);
    }
    __syncthreads();
    // parallel suffix sum (Hillis-Steele), suf[b] = count in bins >= b
    suf[tid] = hist[tid];
    __syncthreads();
    for (int off = 1; off < 256; off <<= 1) {
      unsigned int add = (tid + off < 256) ? suf[tid + off] : 0u;
      __syncthreads();
      suf[tid] += add;
      __syncthreads();
    }
    unsigned int above = suf[tid] - hist[tid];   // strictly above bin tid
    if (above < need && suf[tid] >= need) {      // exactly one bin matches
      misc[0] = prefix | ((unsigned int)tid << shift);
      misc[1] = need - above;
    }
    __syncthreads();
  }

  const unsigned int T      = misc[0];   // exact bits of K-th largest value
  const unsigned int needEq = misc[1];   // how many ==T survive

  for (int i = tid; i < N_DIM; i += 256) {
    unsigned int bv = v[i];
    bool keep = bv > T;
    if (bv == T) {
      unsigned int p = atomicAdd(&misc[2], 1u);
      keep = (p < needEq);
    }
    float fv = keep ? __uint_as_float(bv) : 0.0f;
    z[rowbase + i] = fv;
    if (keep && cidx) {
      unsigned int s = atomicAdd(&misc[3], 1u);
      if (s < K_TOP) {
        cidx[(size_t)row * K_TOP + s] = i;
        cval[(size_t)row * K_TOP + s] = fv;
      }
    }
  }
}

// ---------------------------------------------------------------------------
// W_dec [D,N] -> W_decT [N,D] so decoder gathers are coalesced rows.
// ---------------------------------------------------------------------------
__launch_bounds__(256)
__global__ void wdec_transpose(const float* __restrict__ Wd, float* __restrict__ WT)
{
  __shared__ float t[32][33];
  const int tx = threadIdx.x & 31;
  const int ty = threadIdx.x >> 5;       // 0..7
  const int n0 = blockIdx.x * 32;
  const int d0 = blockIdx.y * 32;
#pragma unroll
  for (int i = 0; i < 4; ++i) {
    int d = d0 + ty + i * 8;
    t[ty + i * 8][tx] = Wd[(size_t)d * N_DIM + n0 + tx];
  }
  __syncthreads();
#pragma unroll
  for (int i = 0; i < 4; ++i) {
    int n = n0 + ty + i * 8;
    WT[(size_t)n * D_DIM + d0 + tx] = t[tx][ty + i * 8];
  }
}

// ---------------------------------------------------------------------------
// Decoder: recon[b,:] = sum_k val_k * Wrow(idx_k) + b.  64 rank-1 updates.
// USE_T: Wsrc = W_decT [N,D] (coalesced). else Wsrc = W_dec [D,N] (L2 gather).
// scan_mode: rebuild (idx,val) by scanning masked z (no-workspace fallback).
// ---------------------------------------------------------------------------
template <bool USE_T>
__launch_bounds__(256)
__global__ void sae_decoder(const float* __restrict__ Wsrc,
                            const int* __restrict__ cidx,
                            const float* __restrict__ cval,
                            const float* __restrict__ z,
                            const float* __restrict__ bias,
                            float* __restrict__ recon,
                            int scan_mode)
{
  __shared__ int          sidx[K_TOP * 2];
  __shared__ float        sval[K_TOP * 2];
  __shared__ unsigned int scnt;
  const int tid = threadIdx.x;
  const int row = blockIdx.x;
  if (tid == 0) scnt = 0u;
  __syncthreads();

  if (!scan_mode) {
    if (tid < K_TOP) {
      sidx[tid] = cidx[(size_t)row * K_TOP + tid];
      sval[tid] = cval[(size_t)row * K_TOP + tid];
    }
  } else {
    for (int i = tid; i < N_DIM; i += 256) {
      float fv = z[(size_t)row * N_DIM + i];
      if (fv != 0.0f) {
        unsigned int s = atomicAdd(&scnt, 1u);
        if (s < K_TOP * 2) { sidx[s] = i; sval[s] = fv; }
      }
    }
  }
  __syncthreads();
  int cnt = K_TOP;
  if (scan_mode) { cnt = (int)scnt; if (cnt > K_TOP * 2) cnt = K_TOP * 2; }

  float a0 = 0.f, a1 = 0.f, a2 = 0.f, a3 = 0.f;
  for (int k = 0; k < cnt; ++k) {
    int   n  = sidx[k];
    float vv = sval[k];
    if (USE_T) {
      const float* wr = Wsrc + (size_t)n * D_DIM;
      a0 += vv * wr[tid];
      a1 += vv * wr[tid + 256];
      a2 += vv * wr[tid + 512];
      a3 += vv * wr[tid + 768];
    } else {
      a0 += vv * Wsrc[(size_t)(tid)       * N_DIM + n];
      a1 += vv * Wsrc[(size_t)(tid + 256) * N_DIM + n];
      a2 += vv * Wsrc[(size_t)(tid + 512) * N_DIM + n];
      a3 += vv * Wsrc[(size_t)(tid + 768) * N_DIM + n];
    }
  }
  const size_t rb = (size_t)row * D_DIM;
  recon[rb + tid      ] = a0 + bias[tid      ];
  recon[rb + tid + 256] = a1 + bias[tid + 256];
  recon[rb + tid + 512] = a2 + bias[tid + 512];
  recon[rb + tid + 768] = a3 + bias[tid + 768];
}

// ---------------------------------------------------------------------------
extern "C" void kernel_launch(void* const* d_in, const int* in_sizes, int n_in,
                              void* d_out, int out_size, void* d_ws, size_t ws_size,
                              hipStream_t stream)
{
  (void)in_sizes; (void)n_in; (void)out_size;
  const float* x    = (const float*)d_in[0];   // [B, D]
  const float* Wenc = (const float*)d_in[1];   // [N, D]
  const float* Wdec = (const float*)d_in[2];   // [D, N]
  const float* bias = (const float*)d_in[3];   // [D]
  // d_in[4] = k (== K_TOP, compile-time)

  float* recon = (float*)d_out;                          // [B, D]
  float* z     = (float*)d_out + (size_t)B_DIM * D_DIM;  // [B, N]

  const size_t cbytes       = (size_t)B_DIM * K_TOP * sizeof(int);
  int*   cidx = (int*)d_ws;
  float* cval = (float*)((char*)d_ws + cbytes);
  float* WT   = (float*)((char*)d_ws + 2 * cbytes);
  const size_t need_compact = 2 * cbytes;                                   // 4 MiB
  const size_t need_T       = need_compact + (size_t)D_DIM * N_DIM * 4;     // +64 MiB
  const bool haveCompact = ws_size >= need_compact;
  const bool haveT       = ws_size >= need_T;

  dim3 egrid(N_DIM / BN, B_DIM / BM);
  sae_encoder<<<egrid, 256, 0, stream>>>(x, Wenc, bias, z);

  const size_t shmem = (size_t)(N_DIM + 256 + 256 + 8) * sizeof(unsigned int);
  sae_topk<<<B_DIM, 256, shmem, stream>>>(z,
                                          haveCompact ? cidx : nullptr,
                                          haveCompact ? cval : nullptr);

  if (haveT) {
    dim3 tgrid(N_DIM / 32, D_DIM / 32);
    wdec_transpose<<<tgrid, 256, 0, stream>>>(Wdec, WT);
    sae_decoder<true><<<B_DIM, 256, 0, stream>>>(WT, cidx, cval, z, bias, recon, 0);
  } else if (haveCompact) {
    sae_decoder<false><<<B_DIM, 256, 0, stream>>>(Wdec, cidx, cval, z, bias, recon, 0);
  } else {
    sae_decoder<false><<<B_DIM, 256, 0, stream>>>(Wdec, nullptr, nullptr, z, bias, recon, 1);
  }
}
